// Model_16449724743715
// MI455X (gfx1250) — matrix operand order repaired and checked by reading, not compiled, not run on hardware
//
#include <hip/hip_runtime.h>
#include <hip/hip_bf16.h>

// ---------------------------------------------------------------------------
// GNN link-predictor pipeline for MI455X (gfx1250, wave32, WMMA).
// Dominant cost: dense cascade ~1.06 TFLOP -> bf16 WMMA (f32 accumulate).
// GEMM: each wave computes a 32x32 C region (2x2 WMMA tiles) -> 2 VMEM
// b128 load per v_wmma (was 4), doubling register-level operand reuse.
// ---------------------------------------------------------------------------

#define N_V 50000
#define N_E 500000
#define N_P 200000
#define NNF 128
#define EMBD 128
#define HDIM 256       // gnn hidden
#define PWF 14
#define DHID 512
#define LDH 2592       // padded row stride for pair features (2574 -> 81*32)
#define BN_EPS 1e-5f

typedef __attribute__((ext_vector_type(16))) __bf16 v16bf;
typedef __attribute__((ext_vector_type(8)))  float  v8f;
typedef __attribute__((ext_vector_type(4)))  unsigned int v4u;

union FragBF { v16bf h; v4u u[2]; };

// ------------------------------ utility -----------------------------------

__global__ void k_zero_f32(float* __restrict__ p, long long n) {
  long long i = (long long)blockIdx.x * blockDim.x + threadIdx.x;
  if (i < n) p[i] = 0.f;
}

__global__ void k_concat_gf(const float* __restrict__ nf,
                            const float* __restrict__ emb,
                            float* __restrict__ gf) {
  int i = blockIdx.x * blockDim.x + threadIdx.x;           // N_V * HDIM
  if (i >= N_V * HDIM) return;
  int r = i >> 8, c = i & 255;
  gf[i] = (c < NNF) ? nf[r * NNF + c] : emb[r * EMBD + (c - NNF)];
}

__global__ void k_deg(const int* __restrict__ dst, float* __restrict__ degcnt) {
  int e = blockIdx.x * blockDim.x + threadIdx.x;
  if (e < N_E) atomicAdd(&degcnt[dst[e]], 1.0f);
}

// one wave per edge: 32 lanes x 8 channels = 256 channel scatter
__global__ void k_edge_agg(const float* __restrict__ gf,
                           const int* __restrict__ src,
                           const int* __restrict__ dst,
                           float* __restrict__ agg) {
  int idx = blockIdx.x * blockDim.x + threadIdx.x;         // N_E * 32
  int e = idx >> 5;
  if (e >= N_E) return;
  int g = idx & 31;
  int s = src[e], d = dst[e];
  const float* in = gf + (size_t)s * HDIM + g * 8;
  float* out = agg + (size_t)d * HDIM + g * 8;
#pragma unroll
  for (int i = 0; i < 8; ++i) atomicAdd(&out[i], in[i]);
}

// A operand for SAGE gemm: [bf16(gf) | bf16(agg/deg)]  -> [N_V, 512]
__global__ void k_make_abuf(const float* __restrict__ gf,
                            const float* __restrict__ agg,
                            const float* __restrict__ degcnt,
                            __bf16* __restrict__ A) {
  int i = blockIdx.x * blockDim.x + threadIdx.x;           // N_V * 512
  if (i >= N_V * 2 * HDIM) return;
  int r = i >> 9, c = i & 511;
  float v = (c < HDIM) ? gf[(size_t)r * HDIM + c]
                       : agg[(size_t)r * HDIM + (c - HDIM)] / fmaxf(degcnt[r], 1.0f);
  A[i] = (__bf16)v;
}

// ---------------------------- weight packing -------------------------------

// Wt[n][k] = k<256 ? Ws[k][n] : Wn[k-256][n]   (bf16, [256,512])
__global__ void k_pack_sage(const float* __restrict__ Ws,
                            const float* __restrict__ Wn,
                            __bf16* __restrict__ Wt) {
  int i = blockIdx.x * blockDim.x + threadIdx.x;           // 256*512
  if (i >= HDIM * 2 * HDIM) return;
  int n = i >> 9, k = i & 511;
  float v = (k < HDIM) ? Ws[k * HDIM + n] : Wn[(k - HDIM) * HDIM + n];
  Wt[i] = (__bf16)v;
}

// dWt[n][k] = k<ci ? dW[k][n] : 0   (bf16, [co, kpad])
__global__ void k_pack_dense(const float* __restrict__ W, __bf16* __restrict__ Wt,
                             int ci, int co, int kpad) {
  int i = blockIdx.x * blockDim.x + threadIdx.x;           // co * kpad
  if (i >= co * kpad) return;
  int n = i / kpad, k = i - n * kpad;
  Wt[i] = (k < ci) ? (__bf16)W[(size_t)k * co + n] : (__bf16)0.f;
}

// ------------------------------- WMMA GEMM ---------------------------------
// C[M,Ncols] = A[M,K](bf16) x Bt[Ncols,K](bf16)^T + bias ; K multiple of 32,
// Ncols multiple of 32. Each wave owns a 32x32 C region = 2x2 WMMA tiles
// (2 b128 loads per v_wmma). 8 waves/block stacked along M. Second M-tile is
// store-guarded with a wave-uniform predicate (A has >=16 pad rows), so EXEC
// stays all-ones through every WMMA as the ISA requires.
// Fragment layouts per CDNA5 ISA 7.12.2 (16-bit A 16x32; B 32x16; f32 C 16x16).

#define WMMA_BF16(A_, B_, C_) \
  __builtin_amdgcn_wmma_f32_16x16x32_bf16(false, (A_), false, (B_), (short)0, (C_), false, false)

__global__ void k_gemm_bf16(const __bf16* __restrict__ A, int lda,
                            const __bf16* __restrict__ Bt, int ldb,
                            const float* __restrict__ bias,
                            float* __restrict__ C, int ldc,
                            int mtiles16, int K) {
  int wid  = threadIdx.x >> 5;
  int lane = threadIdx.x & 31;
  int mt = (blockIdx.y * 8 + wid) * 2;      // first of two 16-row tiles
  if (mt >= mtiles16) return;               // wave-uniform: EXEC stays all-ones
  bool hasM1 = (mt + 1) < mtiles16;         // wave-uniform store guard
  int half = lane >> 4;                     // K-half select
  int r    = lane & 15;
  size_t arow0 = (size_t)(mt * 16 + r) * lda;
  size_t arow1 = arow0 + (size_t)16 * lda;  // pad rows exist if !hasM1
  int n0 = blockIdx.x * 32;
  size_t brow0 = (size_t)(n0 + r) * ldb;
  size_t brow1 = brow0 + (size_t)16 * ldb;
  v8f acc00 = {}, acc01 = {}, acc10 = {}, acc11 = {};
  for (int k0 = 0; k0 < K; k0 += 32) {
    FragBF a0, a1, b0, b1;
    const __bf16* ap0 = A + arow0 + k0 + half * 8;   // K = k0+8h .. +7, then +16
    a0.u[0] = *(const v4u*)(ap0);
    a0.u[1] = *(const v4u*)(ap0 + 16);
    const __bf16* ap1 = A + arow1 + k0 + half * 8;
    a1.u[0] = *(const v4u*)(ap1);
    a1.u[1] = *(const v4u*)(ap1 + 16);
    const __bf16* bp0 = Bt + brow0 + k0 + half * 8;  // K = k0+8h .. +7, then +16
    b0.u[0] = *(const v4u*)(bp0);
    b0.u[1] = *(const v4u*)(bp0 + 16);
    const __bf16* bp1 = Bt + brow1 + k0 + half * 8;
    b1.u[0] = *(const v4u*)(bp1);
    b1.u[1] = *(const v4u*)(bp1 + 16);
    acc00 = WMMA_BF16(a0.h, b0.h, acc00);
    acc01 = WMMA_BF16(a0.h, b1.h, acc01);
    acc10 = WMMA_BF16(a1.h, b0.h, acc10);
    acc11 = WMMA_BF16(a1.h, b1.h, acc11);
  }
  float bv0 = bias[n0 + r];
  float bv1 = bias[n0 + 16 + r];
  int rb0 = mt * 16 + half * 8;             // C: vgpr j -> M = j + 8*half, N = r
#pragma unroll
  for (int j = 0; j < 8; ++j) {
    C[(size_t)(rb0 + j) * ldc + n0 + r]      = acc00[j] + bv0;
    C[(size_t)(rb0 + j) * ldc + n0 + 16 + r] = acc01[j] + bv1;
  }
  if (hasM1) {
    int rb1 = rb0 + 16;
#pragma unroll
    for (int j = 0; j < 8; ++j) {
      C[(size_t)(rb1 + j) * ldc + n0 + r]      = acc10[j] + bv0;
      C[(size_t)(rb1 + j) * ldc + n0 + 16 + r] = acc11[j] + bv1;
    }
  }
}

// ------------------------------ batch norm ---------------------------------

__global__ void k_bn_partial(const float* __restrict__ X, int ldx, int M, int ncols,
                             float* __restrict__ sum, float* __restrict__ sumsq,
                             int rowchunks) {
  __shared__ float ssum[8][32];
  __shared__ float ssq[8][32];
  int t = threadIdx.x;
  int col = blockIdx.x * 32 + (t & 31);
  int sub = t >> 5;
  int rowsPer = (M + rowchunks - 1) / rowchunks;
  int r0 = blockIdx.y * rowsPer;
  int r1 = min(r0 + rowsPer, M);
  float s = 0.f, s2 = 0.f;
  if (col < ncols) {
    for (int r = r0 + sub; r < r1; r += 8) {
      float v = X[(size_t)r * ldx + col];
      s += v; s2 += v * v;
    }
  }
  ssum[sub][t & 31] = s;
  ssq[sub][t & 31]  = s2;
  __syncthreads();
  if (t < 32 && col < ncols) {
    float a = 0.f, b = 0.f;
#pragma unroll
    for (int j = 0; j < 8; ++j) { a += ssum[j][t]; b += ssq[j][t]; }
    atomicAdd(&sum[col], a);
    atomicAdd(&sumsq[col], b);
  }
}

__global__ void k_bn_finalize(const float* __restrict__ sum,
                              const float* __restrict__ sumsq,
                              int M, int ncols,
                              float* __restrict__ mean, float* __restrict__ rstd) {
  int c = blockIdx.x * blockDim.x + threadIdx.x;
  if (c >= ncols) return;
  float mu = sum[c] / (float)M;
  float var = sumsq[c] / (float)M - mu * mu;
  mean[c] = mu;
  rstd[c] = rsqrtf(var + BN_EPS);
}

// gf = (residual ? gf : 0) + relu(bn(y))
__global__ void k_sage_post(const float* __restrict__ y,
                            const float* __restrict__ mean, const float* __restrict__ rstd,
                            const float* __restrict__ g, const float* __restrict__ be,
                            float* __restrict__ gf, int residual) {
  int i = blockIdx.x * blockDim.x + threadIdx.x;           // N_V * HDIM
  if (i >= N_V * HDIM) return;
  int c = i & 255;
  float v = fmaxf((y[i] - mean[c]) * rstd[c] * g[c] + be[c], 0.f);
  gf[i] = (residual ? gf[i] : 0.f) + v;
}

// h[row, 0..525] = [gf[p0] | gf[p1] | pairwise]
__global__ void k_build_pairs(const float* __restrict__ gf,
                              const float* __restrict__ pw,
                              const int* __restrict__ vp,
                              float* __restrict__ h) {
  int row = blockIdx.x;
  int c = blockIdx.y * 256 + threadIdx.x;
  if (c >= 2 * HDIM + PWF) return;
  float v;
  if (c < HDIM)            v = gf[(size_t)vp[row * 2]     * HDIM + c];
  else if (c < 2 * HDIM)   v = gf[(size_t)vp[row * 2 + 1] * HDIM + (c - HDIM)];
  else                     v = pw[(size_t)row * PWF + (c - 2 * HDIM)];
  h[(size_t)row * LDH + c] = v;
}

// xhat = bf16(relu(bn(X)))  with zero pad columns [ncols, kpad)
__global__ void k_bn_norm(const float* __restrict__ X, int ldx,
                          const float* __restrict__ mean, const float* __restrict__ rstd,
                          const float* __restrict__ g, const float* __restrict__ be,
                          int ncols, int kpad, __bf16* __restrict__ xh, int ldxh) {
  int row = blockIdx.x;
  int c = blockIdx.y * 256 + threadIdx.x;
  if (c >= kpad) return;
  float v = 0.f;
  if (c < ncols)
    v = fmaxf((X[(size_t)row * ldx + c] - mean[c]) * rstd[c] * g[c] + be[c], 0.f);
  xh[(size_t)row * ldxh + c] = (__bf16)v;
}

// last layer: co = 1 -> wave-per-row dot product
__global__ void k_final_dot(const __bf16* __restrict__ xh,
                            const float* __restrict__ W,
                            const float* __restrict__ b,
                            float* __restrict__ out) {
  int wid = threadIdx.x >> 5, lane = threadIdx.x & 31;
  int row = blockIdx.x * 8 + wid;
  if (row >= N_P) return;
  const __bf16* xr = xh + (size_t)row * LDH;
  float s = 0.f;
  for (int c = lane; c < 2 * HDIM + PWF + 4 * DHID; c += 32)   // 2574
    s += (float)xr[c] * W[c];
  for (int off = 16; off; off >>= 1) s += __shfl_xor(s, off);
  if (lane == 0) out[row] = s + b[0];
}

// ------------------------------- launcher ----------------------------------

extern "C" void kernel_launch(void* const* d_in, const int* in_sizes, int n_in,
                              void* d_out, int out_size, void* d_ws, size_t ws_size,
                              hipStream_t stream) {
  (void)in_sizes; (void)n_in; (void)out_size; (void)ws_size;
  // params dict flattened leaf-by-leaf in insertion order
  const float* nf  = (const float*)d_in[0];
  const float* pwf = (const float*)d_in[1];
  const float* emb = (const float*)d_in[2];
  const float *Ws[3], *Wn[3], *bs[3], *gs[3], *bes[3];
  for (int l = 0; l < 3; ++l) {
    Ws[l]  = (const float*)d_in[3 + 5 * l];
    Wn[l]  = (const float*)d_in[4 + 5 * l];
    bs[l]  = (const float*)d_in[5 + 5 * l];
    gs[l]  = (const float*)d_in[6 + 5 * l];
    bes[l] = (const float*)d_in[7 + 5 * l];
  }
  const float *dg[5], *dbe[5], *dW[5], *db[5];
  for (int i = 0; i < 5; ++i) {
    dg[i]  = (const float*)d_in[18 + 4 * i];
    dbe[i] = (const float*)d_in[19 + 4 * i];
    dW[i]  = (const float*)d_in[20 + 4 * i];
    db[i]  = (const float*)d_in[21 + 4 * i];
  }
  const int* src = (const int*)d_in[38];
  const int* dst = (const int*)d_in[39];
  const int* vp  = (const int*)d_in[40];
  float* out = (float*)d_out;

  // workspace carve (256B aligned)
  char* w = (char*)d_ws;
  size_t off = 0;
  auto alloc = [&](size_t bytes) -> void* {
    void* p = w + off;
    off += bytes;
    off = (off + 255) & ~(size_t)255;
    return p;
  };
  float*  gf     = (float*)alloc((size_t)N_V * HDIM * 4);
  float*  yb     = (float*)alloc((size_t)N_V * HDIM * 4);
  float*  agg    = (float*)alloc((size_t)N_V * HDIM * 4);
  float*  degcnt = (float*)alloc((size_t)N_V * 4);
  // +16 pad rows: GEMM's second M-tile may speculatively load past N_V rows
  __bf16* Abuf   = (__bf16*)alloc((size_t)(N_V + 16) * 2 * HDIM * 2);
  __bf16* Wt[3];
  for (int l = 0; l < 3; ++l) Wt[l] = (__bf16*)alloc((size_t)HDIM * 2 * HDIM * 2);
  const int ci[4]   = {526, 1038, 1550, 2062};
  const int kpad[4] = {544, 1056, 1568, 2080};
  __bf16* dWt[4];
  for (int i = 0; i < 4; ++i) dWt[i] = (__bf16*)alloc((size_t)DHID * kpad[i] * 2);
  float* bnsum = (float*)alloc(LDH * 4);
  float* bnsq  = (float*)alloc(LDH * 4);
  float* mean  = (float*)alloc(LDH * 4);
  float* rstd  = (float*)alloc(LDH * 4);
  float*  h  = (float*)alloc((size_t)N_P * LDH * 4);
  __bf16* xh = (__bf16*)alloc((size_t)N_P * LDH * 2);

  auto zero = [&](float* p, long long n) {
    k_zero_f32<<<(unsigned)((n + 255) / 256), 256, 0, stream>>>(p, n);
  };

  // ---- weight packing (small) ----
  for (int l = 0; l < 3; ++l)
    k_pack_sage<<<(HDIM * 2 * HDIM + 255) / 256, 256, 0, stream>>>(Ws[l], Wn[l], Wt[l]);
  for (int i = 0; i < 4; ++i)
    k_pack_dense<<<(DHID * kpad[i] + 255) / 256, 256, 0, stream>>>(
        dW[i], dWt[i], ci[i], DHID, kpad[i]);

  // ---- graph features ----
  k_concat_gf<<<(N_V * HDIM + 255) / 256, 256, 0, stream>>>(nf, emb, gf);
  zero(degcnt, N_V);
  k_deg<<<(N_E + 255) / 256, 256, 0, stream>>>(dst, degcnt);

  const int BN_CHUNKS = 120;
  // ---- 3 SAGE layers ----
  for (int l = 0; l < 3; ++l) {
    zero(agg, (long long)N_V * HDIM);
    k_edge_agg<<<(N_E * 32 + 255) / 256, 256, 0, stream>>>(gf, src, dst, agg);
    k_make_abuf<<<(N_V * 2 * HDIM + 255) / 256, 256, 0, stream>>>(gf, agg, degcnt, Abuf);
    {
      int mtiles16 = N_V / 16;                             // 3125
      int groups32 = (mtiles16 + 1) / 2;                   // 1563
      dim3 grid(HDIM / 32, (groups32 + 7) / 8);
      k_gemm_bf16<<<grid, 256, 0, stream>>>(Abuf, 2 * HDIM, Wt[l], 2 * HDIM,
                                            bs[l], yb, HDIM, mtiles16, 2 * HDIM);
    }
    zero(bnsum, LDH); zero(bnsq, LDH);
    k_bn_partial<<<dim3((HDIM + 31) / 32, BN_CHUNKS), 256, 0, stream>>>(
        yb, HDIM, N_V, HDIM, bnsum, bnsq, BN_CHUNKS);
    k_bn_finalize<<<(HDIM + 255) / 256, 256, 0, stream>>>(bnsum, bnsq, N_V, HDIM, mean, rstd);
    k_sage_post<<<(N_V * HDIM + 255) / 256, 256, 0, stream>>>(
        yb, mean, rstd, gs[l], bes[l], gf, l > 0);
  }

  // ---- pair features ----
  k_build_pairs<<<dim3(N_P, (2 * HDIM + PWF + 255) / 256), 256, 0, stream>>>(gf, pwf, vp, h);

  // ---- dense cascade blocks 1..4 (WMMA) ----
  for (int i = 0; i < 4; ++i) {
    zero(bnsum, LDH); zero(bnsq, LDH);
    k_bn_partial<<<dim3((ci[i] + 31) / 32, BN_CHUNKS), 256, 0, stream>>>(
        h, LDH, N_P, ci[i], bnsum, bnsq, BN_CHUNKS);
    k_bn_finalize<<<(ci[i] + 255) / 256, 256, 0, stream>>>(bnsum, bnsq, N_P, ci[i], mean, rstd);
    k_bn_norm<<<dim3(N_P, (kpad[i] + 255) / 256), 256, 0, stream>>>(
        h, LDH, mean, rstd, dg[i], dbe[i], ci[i], kpad[i], xh, LDH);
    int mtiles16 = N_P / 16;                               // 12500
    int groups32 = mtiles16 / 2;                           // 6250
    dim3 grid(DHID / 32, (groups32 + 7) / 8);
    k_gemm_bf16<<<grid, 256, 0, stream>>>(xh, LDH, dWt[i], kpad[i],
                                          db[i], h + ci[i], LDH, mtiles16, kpad[i]);
  }

  // ---- block 5: co = 1 ----
  const int c5 = 2 * HDIM + PWF + 4 * DHID;                // 2574
  zero(bnsum, LDH); zero(bnsq, LDH);
  k_bn_partial<<<dim3((c5 + 31) / 32, BN_CHUNKS), 256, 0, stream>>>(
      h, LDH, N_P, c5, bnsum, bnsq, BN_CHUNKS);
  k_bn_finalize<<<(c5 + 255) / 256, 256, 0, stream>>>(bnsum, bnsq, N_P, c5, mean, rstd);
  k_bn_norm<<<dim3(N_P, (LDH + 255) / 256), 256, 0, stream>>>(
      h, LDH, mean, rstd, dg[4], dbe[4], c5, LDH, xh, LDH);
  k_final_dot<<<(N_P + 7) / 8, 256, 0, stream>>>(xh, dW[4], db[4], out);
}
